// ExactPosteriorECD_67594195304515
// MI455X (gfx1250) — matrix operand order repaired and checked by reading, not compiled, not run on hardware
//
#include <hip/hip_runtime.h>

// ---------------------------------------------------------------------------
// Types for gfx1250 WMMA (wave32) + TDM descriptors
// ---------------------------------------------------------------------------
typedef __attribute__((ext_vector_type(16))) __bf16 v16bf;
typedef __attribute__((ext_vector_type(8)))  float  v8f;
typedef __attribute__((ext_vector_type(4)))  unsigned int v4u;
typedef __attribute__((ext_vector_type(8)))  int    v8i;
typedef __attribute__((ext_vector_type(4)))  int    v4i;

union BF16Frag {
    v16bf        v;
    unsigned int u[8];
};

__device__ __forceinline__ unsigned int f2bf1(float f) {
    unsigned int x = __float_as_uint(f);
    return (x + 0x7FFFu + ((x >> 16) & 1u)) >> 16;  // RNE f32 -> bf16 bits
}
__device__ __forceinline__ unsigned int pk2bf(float lo, float hi) {
    return f2bf1(lo) | (f2bf1(hi) << 16);
}

#define TM  128   // block tile M
#define TN  64    // block tile N
#define TKC 32    // K chunk

// ---------------------------------------------------------------------------
// TDM: DMA a TM x TKC f32 tile (row-major, row stride = strideK elems) into LDS.
// D# per CDNA5 ISA ch.8: group0 {count=1, lds_addr, global_addr, type=2},
// group1 {data_size=4B, tensor_dim0/1 = remaining extents (OOB reads -> 0),
//         tile_dim0=TKC, tile_dim1=TM, tensor_dim0_stride=strideK}.
// Wave-level op (EXEC ignored); caller gates to one wave; TENSORcnt tracks it.
// 6-arg builtin on this toolchain: (v4u, v8i, v4i, v4i, v8i, i32).
// ---------------------------------------------------------------------------
__device__ __forceinline__ void tdm_load_a_tile(unsigned lds_off, const float* gsrc,
                                                int rem_k, int rem_m, int strideK)
{
    unsigned long long ga = (unsigned long long)(size_t)gsrc;
    v4u g0;
    g0[0] = 1u;                                              // count=1, user mode
    g0[1] = lds_off;                                         // LDS byte address
    g0[2] = (unsigned)(ga & 0xFFFFFFFFu);                    // global_addr[31:0]
    g0[3] = (unsigned)((ga >> 32) & 0x1FFFFFFu) | (2u << 30); // addr[56:32] | type=2

    unsigned td0 = (unsigned)rem_k;                          // dim0 extent (elems)
    unsigned td1 = (unsigned)rem_m;                          // dim1 extent (rows)
    unsigned long long s0 = (unsigned long long)(unsigned)strideK;
    v8i g1;
    g1[0] = (int)(2u << 16);                                 // data_size=2 (4 bytes)
    g1[1] = (int)((td0 & 0xFFFFu) << 16);                    // tensor_dim0[15:0]
    g1[2] = (int)((td0 >> 16) | ((td1 & 0xFFFFu) << 16));    // td0[31:16] | td1[15:0]
    g1[3] = (int)((td1 >> 16) | ((unsigned)TKC << 16));      // td1[31:16] | tile_dim0
    g1[4] = (int)(TM & 0xFFFF);                              // tile_dim1 | tile_dim2=0
    g1[5] = (int)(s0 & 0xFFFFFFFFu);                         // dim0_stride[31:0]
    g1[6] = (int)((s0 >> 32) & 0xFFFFu);                     // stride[47:32] | dim1_stride=0
    g1[7] = 0;
    v4i gz4 = {0, 0, 0, 0};
    v8i gz8 = {0, 0, 0, 0, 0, 0, 0, 0};
    __builtin_amdgcn_tensor_load_to_lds(g0, g1, gz4, gz4, gz8, 0);
}

// ---------------------------------------------------------------------------
// GEMM1: Out[M x Ncols] = A[M x K] * W[K x Ncols], bf16 WMMA, f32 accum.
// 256 threads = 8 waves. Block tile 128(M) x 64(N); wave wv owns rows
// [wv*16, wv*16+16) x all 64 cols = 4 WMMA tiles. K chunk = 32.
// A tile DMA'd by TDM (f32 in LDS, double-buffered, OOB zero-fill covers
// both K and M tails). B tile staged as packed bf16x2, double-buffered,
// with uniform full/partial branches only. One barrier per chunk; TDM of
// chunk c+1 overlaps compute of chunk c. Epilogue uses a wave-uniform
// bounds split so the common store path is branch-free.
// ---------------------------------------------------------------------------
__global__ __launch_bounds__(256) void gemm_x_w_bf16(
    const float* __restrict__ A, const float* __restrict__ W,
    float* __restrict__ Out, int M, int K, int Ncols)
{
    __shared__ float        laf[2][TM * TKC];        // f32 A tile   (2 x 16 KB)
    __shared__ unsigned int lbb[2][TN * (TKC / 2)];  // bf16x2 B tile (2 x 4 KB)

    const int tid  = threadIdx.x;
    const int lane = tid & 31;
    const int wv   = tid >> 5;   // 0..7
    const int l16  = lane & 15;
    const int hi   = lane >> 4;

    const int m0 = blockIdx.x * TM;
    const int n0 = blockIdx.y * TN;
    const int nck = (K + TKC - 1) / TKC;

    v8f zero = {0.f, 0.f, 0.f, 0.f, 0.f, 0.f, 0.f, 0.f};
    v8f acc[4] = {zero, zero, zero, zero};

    // ---- B staging: 64 cols x 16 k-pairs = 1024 uints, 4 per thread ----
    auto stage_b = [&](int kk, int buf) {
        bool full = (kk + TKC <= K);   // uniform
#pragma unroll
        for (int i = 0; i < 4; ++i) {
            int idx = tid + 256 * i;
            int col = idx & 63;
            int kp  = idx >> 6;        // 0..15
            int gk  = kk + kp * 2;
            float w0, w1;
            if (full) {
                w0 = W[(size_t)gk * Ncols + n0 + col];
                w1 = W[(size_t)(gk + 1) * Ncols + n0 + col];
            } else {
                w0 = (gk     < K) ? W[(size_t)gk * Ncols + n0 + col] : 0.f;
                w1 = (gk + 1 < K) ? W[(size_t)(gk + 1) * Ncols + n0 + col] : 0.f;
            }
            lbb[buf][col * (TKC / 2) + kp] = pk2bf(w0, w1);
        }
    };

    // ---- prologue: chunk 0 in flight ----
    if (wv == 0)
        tdm_load_a_tile((unsigned)(size_t)&laf[0][0],
                        A + (size_t)m0 * K, K, M - m0, K);
    stage_b(0, 0);
    if (wv == 0) __builtin_amdgcn_s_wait_tensorcnt(0);
    __syncthreads();

    for (int ck = 0; ck < nck; ++ck) {
        const int buf  = ck & 1;
        const int nbuf = buf ^ 1;
        const int kk   = ck * TKC;

        if (ck + 1 < nck) {
            if (wv == 0)
                tdm_load_a_tile((unsigned)(size_t)&laf[nbuf][0],
                                A + (size_t)m0 * K + (kk + TKC),
                                K - (kk + TKC), M - m0, K);
            stage_b(kk + TKC, nbuf);
            // keep next W rows warm (emits global_prefetch_b8)
            __builtin_prefetch(W + (size_t)(kk + TKC + (tid >> 6) * 8) * Ncols + n0 + (tid & 63), 0, 0);
        }

        // ---- compute chunk ck ----
        // A fragment 16x32: lane l16 = row; K phases per ISA 7.12.2 layout
        BF16Frag af;
        const float2* ap = (const float2*)&laf[buf][(wv * 16 + l16) * TKC];
#pragma unroll
        for (int vv = 0; vv < 4; ++vv) {
            float2 p0 = ap[hi * 4 + vv];       // K = hi*8 + 2v
            float2 p1 = ap[8 + hi * 4 + vv];   // K = 16 + hi*8 + 2v
            af.u[vv]     = pk2bf(p0.x, p0.y);
            af.u[vv + 4] = pk2bf(p1.x, p1.y);
        }
#pragma unroll
        for (int ni = 0; ni < 4; ++ni) {
            BF16Frag bfv;
            const unsigned int* bp = &lbb[buf][(ni * 16 + l16) * (TKC / 2)];
#pragma unroll
            for (int vv = 0; vv < 8; ++vv)
                bfv.u[vv] = bp[(vv >> 2) * 8 + hi * 4 + (vv & 3)];   // K = (v/4)*16 + hi*8 + 2(v%4)
            acc[ni] = __builtin_amdgcn_wmma_f32_16x16x32_bf16(
                false, af.v, false, bfv.v, (short)0, acc[ni], false, false);
        }

        if (wv == 0) __builtin_amdgcn_s_wait_tensorcnt(0);
        __syncthreads();
    }

    // ---- epilogue: C/D layout VGPR v -> M = v + hi*8, N = l16 ----
    const int rbase = m0 + wv * 16;                 // wave's 16-row strip
    float* obase = Out + (size_t)(rbase + hi * 8) * Ncols + n0 + l16;
    if (rbase + 16 <= M) {
        // common case: whole strip in bounds, branch-free immediate-offset stores
#pragma unroll
        for (int ni = 0; ni < 4; ++ni)
#pragma unroll
            for (int vv = 0; vv < 8; ++vv)
                obase[(size_t)vv * Ncols + ni * 16] = acc[ni][vv];
    } else if (rbase < M) {
        // partial strip (last block only)
#pragma unroll
        for (int ni = 0; ni < 4; ++ni)
#pragma unroll
            for (int vv = 0; vv < 8; ++vv)
                if (rbase + hi * 8 + vv < M)
                    obase[(size_t)vv * Ncols + ni * 16] = acc[ni][vv];
    }
}

// ---------------------------------------------------------------------------
// GEMM2: Out[M x 16] = A[M x K] * W[K x 16]; one wave per 16x16 output tile.
// K is a multiple of 32 here (1024). W tiny -> L2 resident.
// ---------------------------------------------------------------------------
__global__ __launch_bounds__(32) void gemm_h_w2_bf16(
    const float* __restrict__ A, const float* __restrict__ W,
    float* __restrict__ Out, int M, int K)
{
    const int lane = threadIdx.x & 31;
    const int l16  = lane & 15;
    const int hi   = lane >> 4;
    const int m0   = blockIdx.x * 16;

    int row = m0 + l16; if (row >= M) row = M - 1;
    const float* ar = A + (size_t)row * K;

    v8f acc = {0.f, 0.f, 0.f, 0.f, 0.f, 0.f, 0.f, 0.f};

    for (int kk = 0; kk < K; kk += 32) {
        BF16Frag af, bfv;
#pragma unroll
        for (int vv = 0; vv < 4; ++vv) {
            int k0 = kk + hi * 8 + vv * 2;
            af.u[vv]     = pk2bf(ar[k0],      ar[k0 + 1]);
            af.u[vv + 4] = pk2bf(ar[k0 + 16], ar[k0 + 17]);
        }
#pragma unroll
        for (int vv = 0; vv < 8; ++vv) {
            int k = kk + (vv >> 2) * 16 + hi * 8 + (vv & 3) * 2;
            bfv.u[vv] = pk2bf(W[(size_t)k * 16 + l16], W[(size_t)(k + 1) * 16 + l16]);
        }
        acc = __builtin_amdgcn_wmma_f32_16x16x32_bf16(
            false, af.v, false, bfv.v, (short)0, acc, false, false);
    }

    float* ob = Out + (size_t)(m0 + hi * 8) * 16 + l16;
    if (m0 + 16 <= M) {
#pragma unroll
        for (int vv = 0; vv < 8; ++vv) ob[vv * 16] = acc[vv];
    } else {
#pragma unroll
        for (int vv = 0; vv < 8; ++vv)
            if (m0 + hi * 8 + vv < M) ob[vv * 16] = acc[vv];
    }
}

// ---------------------------------------------------------------------------
// Graph-norm + aggregation + activation + fusion kernels
// ---------------------------------------------------------------------------
__global__ void deg_init_k(float* deg, int n) {
    int i = blockIdx.x * 256 + threadIdx.x;
    if (i < n) deg[i] = 1.0f;  // self-loop
}

__global__ void deg_accum_k(const int* __restrict__ dst, float* deg, int e) {
    int i = blockIdx.x * 256 + threadIdx.x;
    if (i < e) atomicAdd(&deg[dst[i]], 1.0f);
}

__global__ void dinv_kernel(const float* __restrict__ deg, float* __restrict__ dinv, int n) {
    int i = blockIdx.x * 256 + threadIdx.x;
    if (i < n) { float d = deg[i]; dinv[i] = (d > 0.f) ? rsqrtf(d) : 0.f; }
}

// agg[i,j] = dinv[i]^2 * h[i,j] + bias[j]   (self-loop term + bias)
__global__ void agg_init_k(const float* __restrict__ h, const float* __restrict__ dinv,
                           const float* __restrict__ bias, float* __restrict__ agg,
                           int total, int shift, int mask) {
    int i = blockIdx.x * 256 + threadIdx.x;
    if (i < total) {
        int row = i >> shift;
        int j   = i & mask;
        float di = dinv[row];
        agg[i] = di * di * h[i] + bias[j];
    }
}

// one block per edge; 256 threads x float4 = 1024 features
__global__ __launch_bounds__(256) void edge_scatter_hid_k(
    const float* __restrict__ h, const int* __restrict__ src, const int* __restrict__ dst,
    const float* __restrict__ dinv, float* __restrict__ agg, int hid)
{
    int e = blockIdx.x;
    int s = src[e], d = dst[e];
    float nm = dinv[s] * dinv[d];
    const float4* hp = (const float4*)(h + (size_t)s * hid);
    float*        ap = agg + (size_t)d * hid;
    int j = threadIdx.x;
    float4 vv = hp[j];
    atomicAdd(ap + 4 * j + 0, nm * vv.x);
    atomicAdd(ap + 4 * j + 1, nm * vv.y);
    atomicAdd(ap + 4 * j + 2, nm * vv.z);
    atomicAdd(ap + 4 * j + 3, nm * vv.w);
}

__global__ void relu_k(float* __restrict__ x, int n) {
    int i = blockIdx.x * 256 + threadIdx.x;
    if (i < n) x[i] = fmaxf(x[i], 0.f);
}

// thread per (edge, channel)
__global__ void edge_scatter_c_k(
    const float* __restrict__ g, const int* __restrict__ src, const int* __restrict__ dst,
    const float* __restrict__ dinv, float* __restrict__ agg, int e)
{
    int i = blockIdx.x * 256 + threadIdx.x;
    if (i >= e * 16) return;
    int ei = i >> 4, c = i & 15;
    int s = src[ei], d = dst[ei];
    float nm = dinv[s] * dinv[d];
    atomicAdd(&agg[(size_t)d * 16 + c], nm * g[(size_t)s * 16 + c]);
}

__global__ void softmax16_k(const float* __restrict__ in, float* __restrict__ out, int n) {
    int i = blockIdx.x * 256 + threadIdx.x;
    if (i >= n) return;
    const float* r = in + (size_t)i * 16;
    float m = r[0];
#pragma unroll
    for (int c = 1; c < 16; ++c) m = fmaxf(m, r[c]);
    float ev[16]; float s = 0.f;
#pragma unroll
    for (int c = 0; c < 16; ++c) { ev[c] = __expf(r[c] - m); s += ev[c]; }
    float inv = 1.f / s;
    float* o = out + (size_t)i * 16;
#pragma unroll
    for (int c = 0; c < 16; ++c) o[c] = ev[c] * inv;
}

__global__ void sigmoid16_k(const float* __restrict__ in, float* __restrict__ out, int n) {
    int i = blockIdx.x * 256 + threadIdx.x;
    if (i >= n) return;
    const float* r = in + (size_t)i * 16;
    float* o = out + (size_t)i * 16;
#pragma unroll
    for (int c = 0; c < 16; ++c) o[c] = 1.f / (1.f + __expf(-r[c]));
}

// eta = tanh(eta_param); a = |eta|; alpha = 10a + 2(1-a); prior = alpha/sum
__global__ void eta_prior_k(const float* __restrict__ ep, float* __restrict__ a_ws,
                            float* __restrict__ prior_ws, float* __restrict__ eta_out) {
    if (threadIdx.x == 0 && blockIdx.x == 0) {
        float al[16]; float s = 0.f;
        for (int c = 0; c < 16; ++c) {
            float e = tanhf(ep[c]);
            eta_out[c] = e;
            float a = fabsf(e);
            a_ws[c] = a;
            al[c] = 10.f * a + 2.f * (1.f - a);
            s += al[c];
        }
        float inv = 1.f / s;
        for (int c = 0; c < 16; ++c) prior_ws[c] = al[c] * inv;
    }
}

__global__ __launch_bounds__(256) void pair_k(
    const float* __restrict__ theta, const float* __restrict__ phi,
    const int* __restrict__ u, const int* __restrict__ v,
    const float* __restrict__ a_ws, const float* __restrict__ prior_ws,
    float* __restrict__ qout, float* __restrict__ pout, int bn)
{
    __shared__ float sa[16], sp[16];
    if (threadIdx.x < 16)       sa[threadIdx.x]      = a_ws[threadIdx.x];
    else if (threadIdx.x < 32)  sp[threadIdx.x - 16] = prior_ws[threadIdx.x - 16];
    __syncthreads();

    int i = blockIdx.x * 256 + threadIdx.x;
    if (i >= bn) return;
    int uu = u[i], vv = v[i];

    float tu_[16], tv_[16], pu_[16], pv_[16];
    const float4* tu4 = (const float4*)(theta + (size_t)uu * 16);
    const float4* tv4 = (const float4*)(theta + (size_t)vv * 16);
    const float4* pu4 = (const float4*)(phi   + (size_t)uu * 16);
    const float4* pv4 = (const float4*)(phi   + (size_t)vv * 16);
#pragma unroll
    for (int w = 0; w < 4; ++w) {
        ((float4*)tu_)[w] = tu4[w];
        ((float4*)tv_)[w] = tv4[w];
        ((float4*)pu_)[w] = pu4[w];
        ((float4*)pv_)[w] = pv4[w];
    }

    float p[16], q[16]; float qs = 0.f;
#pragma unroll
    for (int c = 0; c < 16; ++c) {
        float ac = sa[c];
        float pp = ac * tu_[c] * tv_[c] + (1.f - ac) * pu_[c] * pv_[c] + 1e-10f;
        p[c] = pp;
        float qq = pp * sp[c];
        q[c] = qq; qs += qq;
    }
    float inv = 1.f / qs;
    float* qo = qout + (size_t)i * 16;
    float* po = pout + (size_t)i * 16;
#pragma unroll
    for (int c = 0; c < 16; ++c) { qo[c] = q[c] * inv; po[c] = p[c]; }
}

// ---------------------------------------------------------------------------
// Host launcher
// ---------------------------------------------------------------------------
extern "C" void kernel_launch(void* const* d_in, const int* in_sizes, int n_in,
                              void* d_out, int out_size, void* d_ws, size_t ws_size,
                              hipStream_t stream)
{
    (void)in_sizes; (void)n_in; (void)out_size; (void)ws_size;
    const int N = 10000, C = 16, E = 640000, B = 2000000, H = 1024;

    const float* X   = (const float*)d_in[0];
    const int*   ei  = (const int*)  d_in[1];
    const int*   u   = (const int*)  d_in[2];
    const int*   v   = (const int*)  d_in[3];
    const float* tW1 = (const float*)d_in[4];
    const float* tb1 = (const float*)d_in[5];
    const float* tW2 = (const float*)d_in[6];
    const float* tb2 = (const float*)d_in[7];
    const float* pW1 = (const float*)d_in[8];
    const float* pb1 = (const float*)d_in[9];
    const float* pW2 = (const float*)d_in[10];
    const float* pb2 = (const float*)d_in[11];
    const float* ep  = (const float*)d_in[12];
    const int* src  = ei;
    const int* dstp = ei + E;

    char* wsb = (char*)d_ws; size_t off = 0;
    auto carve = [&](size_t bytes) -> float* {
        float* p = (float*)(wsb + off);
        off = (off + bytes + 255) & ~(size_t)255;
        return p;
    };
    float* deg   = carve((size_t)N * 4);
    float* dinv  = carve((size_t)N * 4);
    float* a_ws  = carve(64);
    float* pr_ws = carve(64);
    float* h     = carve((size_t)N * H * 4);
    float* agg   = carve((size_t)N * H * 4);
    float* g     = carve((size_t)N * C * 4);
    float* agg2  = carve((size_t)N * C * 4);
    float* theta = carve((size_t)N * C * 4);
    float* phi   = carve((size_t)N * C * 4);

    float* qout = (float*)d_out;
    float* pout = qout + (size_t)B * C;
    float* eout = pout + (size_t)B * C;

    deg_init_k <<<(N + 255) / 256, 256, 0, stream>>>(deg, N);
    deg_accum_k<<<(E + 255) / 256, 256, 0, stream>>>(dstp, deg, E);
    dinv_kernel<<<(N + 255) / 256, 256, 0, stream>>>(deg, dinv, N);

    auto branch = [&](const float* W1, const float* b1, const float* W2, const float* b2,
                      float* outNC, bool use_softmax) {
        dim3 g1((N + TM - 1) / TM, H / TN);
        gemm_x_w_bf16<<<g1, 256, 0, stream>>>(X, W1, h, N, N, H);
        agg_init_k<<<((N * H) + 255) / 256, 256, 0, stream>>>(h, dinv, b1, agg, N * H, 10, H - 1);
        edge_scatter_hid_k<<<E, 256, 0, stream>>>(h, src, dstp, dinv, agg, H);
        relu_k<<<((N * H) + 255) / 256, 256, 0, stream>>>(agg, N * H);
        gemm_h_w2_bf16<<<N / 16, 32, 0, stream>>>(agg, W2, g, N, H);
        agg_init_k<<<((N * C) + 255) / 256, 256, 0, stream>>>(g, dinv, b2, agg2, N * C, 4, C - 1);
        edge_scatter_c_k<<<((E * 16) + 255) / 256, 256, 0, stream>>>(g, src, dstp, dinv, agg2, E);
        if (use_softmax) softmax16_k<<<(N + 255) / 256, 256, 0, stream>>>(agg2, outNC, N);
        else             sigmoid16_k<<<(N + 255) / 256, 256, 0, stream>>>(agg2, outNC, N);
    };
    branch(tW1, tb1, tW2, tb2, theta, true);
    branch(pW1, pb1, pW2, pb2, phi,   false);

    eta_prior_k<<<1, 32, 0, stream>>>(ep, a_ws, pr_ws, eout);
    pair_k<<<(B + 255) / 256, 256, 0, stream>>>(theta, phi, u, v, a_ws, pr_ws, qout, pout, B);
}
